// MST_GNN_Layer_86423331930149
// MI455X (gfx1250) — matrix operand order repaired and checked by reading, not compiled, not run on hardware
//
#include <hip/hip_runtime.h>
#include <hip/hip_bf16.h>

// ---------------- types for WMMA ----------------
typedef __attribute__((ext_vector_type(16))) __bf16 v16bf;
typedef __attribute__((ext_vector_type(8)))  float  v8f;

union Frag16 {
    v16bf v;
    unsigned short u[16];
};

__device__ __forceinline__ unsigned short f2bf(float f) {
    union { float f; unsigned int u; } c;
    c.f = f;
    unsigned int u = c.u;
    unsigned int rnd = 0x7FFFu + ((u >> 16) & 1u);   // round-to-nearest-even
    return (unsigned short)((u + rnd) >> 16);
}

__device__ __forceinline__ float sigmoidf_(float x) {
    return 1.0f / (1.0f + __expf(-x));
}

// ---------------- zero ----------------
__global__ void zero_f32(float* __restrict__ p, size_t n) {
    size_t i = (size_t)blockIdx.x * blockDim.x + threadIdx.x;
    size_t st = (size_t)gridDim.x * blockDim.x;
    for (; i < n; i += st) p[i] = 0.0f;
}

// ---------------- degree accumulation ----------------
__global__ void degree_pair(const int* __restrict__ src, const int* __restrict__ dst,
                            float* __restrict__ degOut, float* __restrict__ degIn, int nE) {
    int e = blockIdx.x * blockDim.x + threadIdx.x;
    if (e < nE) {
        atomicAdd(&degOut[src[e]], 1.0f);
        atomicAdd(&degIn[dst[e]], 1.0f);
    }
}

// ---------------- bf16 WMMA GEMM: C[M,128] (=|+=) A[M,128] x B[128,128] ----------------
// B view: transB==0 -> B[k*ldb + n] ; transB==1 -> B[n*ldb + k]
// Optional per-row scale on A: rsqrt(max(deg,1)).
#define GEMM_BM 64
__global__ __launch_bounds__(256) void gemm128_bf16(
    const float* __restrict__ A, int lda,
    const float* __restrict__ rowDeg,     // nullable
    const float* __restrict__ B, int ldb, int transB,
    float* __restrict__ C, int M, int accumulate)
{
    __shared__ __align__(16) unsigned short lsA[GEMM_BM][128];   // [m][k]  16KB
    __shared__ __align__(16) unsigned short lsB[128][128];       // [n][k]  32KB

    const int tid = threadIdx.x;
    const int m0  = blockIdx.x * GEMM_BM;

    // stage B (transposed into [n][k]): 16384 elems / 256 thr = 64 each
    {
        int n  = tid >> 1;
        int kb = (tid & 1) * 64;
        for (int k = kb; k < kb + 64; ++k) {
            float v = transB ? B[n * ldb + k] : B[k * ldb + n];
            lsB[n][k] = f2bf(v);
        }
    }
    // stage A with row scaling: 64x128 / 256 thr = 32 each
    {
        int r  = tid >> 2;
        int kb = (tid & 3) * 32;
        int gm = m0 + r;
        bool ok = gm < M;
        float s = 1.0f;
        if (ok && rowDeg) {
            float d = rowDeg[gm];
            d = d < 1.0f ? 1.0f : d;
            s = rsqrtf(d);
        }
        for (int k = kb; k < kb + 32; ++k) {
            float v = ok ? A[(size_t)gm * lda + k] * s : 0.0f;
            lsA[r][k] = f2bf(v);
        }
    }
    __syncthreads();

    const int wave   = tid >> 5;          // 0..7
    const int lane   = tid & 31;
    const int rstrip = wave & 3;          // rows  rstrip*16
    const int cgroup = wave >> 2;         // cols  cgroup*64 (4 tiles of 16)
    const int half   = lane >> 4;         // 0/1
    const int lmod   = lane & 15;

    v8f acc[4];
    for (int t = 0; t < 4; ++t) acc[t] = (v8f){0,0,0,0,0,0,0,0};

    const int am = rstrip * 16 + lmod;

    for (int kk = 0; kk < 4; ++kk) {
        // A fragment: e[0..15] <- k = kk*32 + half*16 + j
        Frag16 af;
        int ak = kk * 32 + half * 16;
        ((uint4*)af.u)[0] = *(const uint4*)&lsA[am][ak];
        ((uint4*)af.u)[1] = *(const uint4*)&lsA[am][ak + 8];
        for (int t = 0; t < 4; ++t) {
            // B fragment: lane<16 -> n=lmod,k=kk*32+j ; lane>=16 -> k offset +16
            Frag16 bfr;
            int bn = (cgroup * 4 + t) * 16 + lmod;
            int bk = kk * 32 + half * 16;
            ((uint4*)bfr.u)[0] = *(const uint4*)&lsB[bn][bk];
            ((uint4*)bfr.u)[1] = *(const uint4*)&lsB[bn][bk + 8];
            acc[t] = __builtin_amdgcn_wmma_f32_16x16x32_bf16(
                false, af.v, false, bfr.v, (short)0, acc[t], false, false);
        }
    }

    // store C tiles: VGPR i, lane -> row = i + (lane<16?0:8), col = lane%16
    for (int t = 0; t < 4; ++t) {
        int cb = (cgroup * 4 + t) * 16 + lmod;
        for (int i = 0; i < 8; ++i) {
            int gm = m0 + rstrip * 16 + i + half * 8;
            if (gm < M) {
                float v = acc[t][i];
                size_t idx = (size_t)gm * 128 + cb;
                if (accumulate) v += C[idx];
                C[idx] = v;
            }
        }
    }
}

// ---------------- edge scatter with in-degree normalization ----------------
__global__ __launch_bounds__(128) void scatter_edges(
    const float* __restrict__ xs, const int* __restrict__ src, const int* __restrict__ dst,
    const float* __restrict__ degIn, float* __restrict__ sp, int nE)
{
    int e = blockIdx.x;
    if (e >= nE) return;
    int s = src[e];
    int d = dst[e];
    float dg = degIn[d];
    dg = dg < 1.0f ? 1.0f : dg;
    float sc = rsqrtf(dg);
    int f = threadIdx.x;   // 128 features
    atomicAdd(&sp[(size_t)d * 128 + f], xs[(size_t)s * 128 + f] * sc);
}

// ---------------- bias add (1 or 2 relation biases per dst type) ----------------
__global__ void bias_add(float* __restrict__ sp, const float* __restrict__ b1,
                         const float* __restrict__ b2, size_t total)
{
    size_t i = (size_t)blockIdx.x * blockDim.x + threadIdx.x;
    if (i < total) {
        int f = (int)(i & 127);
        float v = sp[i] + b1[f];
        if (b2) v += b2[f];
        sp[i] = v;
    }
}

// ---------------- gate sigmoid (in-place in d_out) + fused state ----------------
__global__ void gate_fuse(float* __restrict__ gateL, const float* __restrict__ gate_b,
                          const float* __restrict__ h1, const float* __restrict__ h2,
                          float* __restrict__ fused, size_t total)
{
    size_t i = (size_t)blockIdx.x * blockDim.x + threadIdx.x;
    if (i < total) {
        int f = (int)(i & 127);
        float g = sigmoidf_(gateL[i] + gate_b[f]);
        gateL[i] = g;
        fused[i] = g * h1[i] + (1.0f - g) * h2[i];
    }
}

// ---------------- final GRU combine + relu ----------------
__global__ void gru_final(const float* __restrict__ rb, const float* __restrict__ zb,
                          const float* __restrict__ n1, const float* __restrict__ n2,
                          const float* __restrict__ fused,
                          const float* __restrict__ b_ih, const float* __restrict__ b_hh,
                          float* __restrict__ out, size_t total)
{
    size_t i = (size_t)blockIdx.x * blockDim.x + threadIdx.x;
    if (i < total) {
        int f = (int)(i & 127);
        float r  = sigmoidf_(rb[i] + b_ih[f]       + b_hh[f]);
        float z  = sigmoidf_(zb[i] + b_ih[128 + f] + b_hh[128 + f]);
        float nn = tanhf((n1[i] + b_ih[256 + f]) + r * (n2[i] + b_hh[256 + f]));
        float o  = (1.0f - z) * nn + z * fused[i];
        out[i] = o > 0.0f ? o : 0.0f;
    }
}

// ==================================================================
extern "C" void kernel_launch(void* const* d_in, const int* in_sizes, int n_in,
                              void* d_out, int out_size, void* d_ws, size_t ws_size,
                              hipStream_t stream) {
    (void)in_sizes; (void)n_in; (void)out_size; (void)ws_size;
    const int NC = 2000, NO = 100000, NE = 200000, D = 128;
    const int NT = NC + NO + NE;                     // 302000
    const int ECO = 100000, EOC = 100000, EEO = 200000, EOE = 200000;

    const float* h_course  = (const float*)d_in[0];
    const float* h_object  = (const float*)d_in[1];
    const float* h_enroll  = (const float*)d_in[2];
    const float* h1_course = (const float*)d_in[3];
    const float* h1_object = (const float*)d_in[4];
    const float* h1_enroll = (const float*)d_in[5];
    const float* h2_course = (const float*)d_in[6];
    const float* h2_object = (const float*)d_in[7];
    const float* h2_enroll = (const float*)d_in[8];
    const int* src_co = (const int*)d_in[9];
    const int* dst_co = (const int*)d_in[10];
    const int* src_oc = (const int*)d_in[11];
    const int* dst_oc = (const int*)d_in[12];
    const int* src_eo = (const int*)d_in[13];
    const int* dst_eo = (const int*)d_in[14];
    const int* src_oe = (const int*)d_in[15];
    const int* dst_oe = (const int*)d_in[16];
    const float* W_co = (const float*)d_in[17];
    const float* b_co = (const float*)d_in[18];
    const float* W_oc = (const float*)d_in[19];
    const float* b_oc = (const float*)d_in[20];
    const float* W_eo = (const float*)d_in[21];
    const float* b_eo = (const float*)d_in[22];
    const float* W_oe = (const float*)d_in[23];
    const float* b_oe = (const float*)d_in[24];
    const float* gate_W = (const float*)d_in[25];    // [128, 384]
    const float* gate_b = (const float*)d_in[26];
    const float* W_ih = (const float*)d_in[27];      // [384, 128]
    const float* W_hh = (const float*)d_in[28];      // [384, 128]
    const float* b_ih = (const float*)d_in[29];
    const float* b_hh = (const float*)d_in[30];

    float* ws = (float*)d_ws;
    size_t p = 0;
    auto alloc = [&](size_t n) { float* r = ws + p; p += n; return r; };

    float* degCoOut = alloc(NC);
    float* degCoIn  = alloc(NO);
    float* degOcOut = alloc(NO);
    float* degOcIn  = alloc(NC);
    float* degEoOut = alloc(NE);
    float* degEoIn  = alloc(NO);
    float* degOeOut = alloc(NO);
    float* degOeIn  = alloc(NE);
    float* sp       = alloc((size_t)NT * D);         // sp_course | sp_object | sp_enroll
    const size_t zeroSpan = p;                       // degrees + sp must be zeroed each call
    float* xs       = alloc((size_t)(NC + 2 * NO + NE) * D);  // xs_co|xs_oc|xs_oe|xs_eo
    float* fused    = alloc((size_t)NT * D);
    float* rbuf     = alloc((size_t)NT * D);
    float* zbuf     = alloc((size_t)NT * D);
    float* n2buf    = alloc((size_t)NT * D);
    float* n1buf    = xs;                            // xs region is free after scatter stage

    float* sp_c = sp;
    float* sp_o = sp + (size_t)NC * D;
    float* sp_e = sp + (size_t)(NC + NO) * D;
    float* xs_co = xs;
    float* xs_oc = xs + (size_t)NC * D;
    float* xs_oe = xs + (size_t)(NC + NO) * D;
    float* xs_eo = xs + (size_t)(NC + 2 * NO) * D;

    float* out  = (float*)d_out;                     // out_c|out_o|out_e
    float* gate = out + (size_t)NT * D;              // g_c|g_o|g_e (contiguous, same row order)

    const size_t totAll = (size_t)NT * D;
    const int EB = 256;
    dim3 blk(256);

    // ---- zero degree + sp accumulators ----
    zero_f32<<<2048, 256, 0, stream>>>(ws, zeroSpan);

    // ---- degrees ----
    degree_pair<<<(ECO + EB - 1) / EB, EB, 0, stream>>>(src_co, dst_co, degCoOut, degCoIn, ECO);
    degree_pair<<<(EOC + EB - 1) / EB, EB, 0, stream>>>(src_oc, dst_oc, degOcOut, degOcIn, EOC);
    degree_pair<<<(EEO + EB - 1) / EB, EB, 0, stream>>>(src_eo, dst_eo, degEoOut, degEoIn, EEO);
    degree_pair<<<(EOE + EB - 1) / EB, EB, 0, stream>>>(src_oe, dst_oe, degOeOut, degOeIn, EOE);

    // ---- xs = (x_src * deg_out^-1/2) @ W  (4 WMMA GEMMs) ----
    auto gblocks = [](int M) { return dim3((M + GEMM_BM - 1) / GEMM_BM); };
    gemm128_bf16<<<gblocks(NC), blk, 0, stream>>>(h_course, D, degCoOut, W_co, D, 0, xs_co, NC, 0);
    gemm128_bf16<<<gblocks(NO), blk, 0, stream>>>(h_object, D, degOcOut, W_oc, D, 0, xs_oc, NO, 0);
    gemm128_bf16<<<gblocks(NO), blk, 0, stream>>>(h_object, D, degOeOut, W_oe, D, 0, xs_oe, NO, 0);
    gemm128_bf16<<<gblocks(NE), blk, 0, stream>>>(h_enroll, D, degEoOut, W_eo, D, 0, xs_eo, NE, 0);

    // ---- scatter with deg_in^-1/2 (object accumulates both relations) ----
    scatter_edges<<<ECO, 128, 0, stream>>>(xs_co, src_co, dst_co, degCoIn, sp_o, ECO);
    scatter_edges<<<EEO, 128, 0, stream>>>(xs_eo, src_eo, dst_eo, degEoIn, sp_o, EEO);
    scatter_edges<<<EOC, 128, 0, stream>>>(xs_oc, src_oc, dst_oc, degOcIn, sp_c, EOC);
    scatter_edges<<<EOE, 128, 0, stream>>>(xs_oe, src_oe, dst_oe, degOeIn, sp_e, EOE);

    // ---- biases per dst type ----
    bias_add<<<((size_t)NC * D + EB - 1) / EB, EB, 0, stream>>>(sp_c, b_oc, nullptr, (size_t)NC * D);
    bias_add<<<((size_t)NO * D + EB - 1) / EB, EB, 0, stream>>>(sp_o, b_co, b_eo, (size_t)NO * D);
    bias_add<<<((size_t)NE * D + EB - 1) / EB, EB, 0, stream>>>(sp_e, b_oe, nullptr, (size_t)NE * D);

    // ---- gate linear: concat[h_sp,h1,h2] @ gate_W.T  (7 WMMA GEMMs into d_out gate region) ----
    // chunk g: B[k][n] = gate_W[n][g*128+k]  -> transB view with ldb=384, ptr offset g*128
    gemm128_bf16<<<gblocks(NT), blk, 0, stream>>>(sp,        D, nullptr, gate_W,       3 * D, 1, gate, NT, 0);
    gemm128_bf16<<<gblocks(NC), blk, 0, stream>>>(h1_course, D, nullptr, gate_W + 128, 3 * D, 1, gate,                        NC, 1);
    gemm128_bf16<<<gblocks(NO), blk, 0, stream>>>(h1_object, D, nullptr, gate_W + 128, 3 * D, 1, gate + (size_t)NC * D,       NO, 1);
    gemm128_bf16<<<gblocks(NE), blk, 0, stream>>>(h1_enroll, D, nullptr, gate_W + 128, 3 * D, 1, gate + (size_t)(NC + NO) * D, NE, 1);
    gemm128_bf16<<<gblocks(NC), blk, 0, stream>>>(h2_course, D, nullptr, gate_W + 256, 3 * D, 1, gate,                        NC, 1);
    gemm128_bf16<<<gblocks(NO), blk, 0, stream>>>(h2_object, D, nullptr, gate_W + 256, 3 * D, 1, gate + (size_t)NC * D,       NO, 1);
    gemm128_bf16<<<gblocks(NE), blk, 0, stream>>>(h2_enroll, D, nullptr, gate_W + 256, 3 * D, 1, gate + (size_t)(NC + NO) * D, NE, 1);

    // ---- gate sigmoid (in place in d_out) + fused state ----
    gate_fuse<<<((size_t)NC * D + EB - 1) / EB, EB, 0, stream>>>(gate,                         gate_b, h1_course, h2_course, fused,                         (size_t)NC * D);
    gate_fuse<<<((size_t)NO * D + EB - 1) / EB, EB, 0, stream>>>(gate + (size_t)NC * D,        gate_b, h1_object, h2_object, fused + (size_t)NC * D,        (size_t)NO * D);
    gate_fuse<<<((size_t)NE * D + EB - 1) / EB, EB, 0, stream>>>(gate + (size_t)(NC + NO) * D, gate_b, h1_enroll, h2_enroll, fused + (size_t)(NC + NO) * D, (size_t)NE * D);

    // ---- GRU gates: chunk g of x@W_ih.T uses W_ih rows [g*128, g*128+128) -> transB ldb=128 ----
    const size_t WC = (size_t)128 * 128;
    gemm128_bf16<<<gblocks(NT), blk, 0, stream>>>(sp,    D, nullptr, W_ih,          D, 1, rbuf,  NT, 0);
    gemm128_bf16<<<gblocks(NT), blk, 0, stream>>>(fused, D, nullptr, W_hh,          D, 1, rbuf,  NT, 1);
    gemm128_bf16<<<gblocks(NT), blk, 0, stream>>>(sp,    D, nullptr, W_ih + WC,     D, 1, zbuf,  NT, 0);
    gemm128_bf16<<<gblocks(NT), blk, 0, stream>>>(fused, D, nullptr, W_hh + WC,     D, 1, zbuf,  NT, 1);
    gemm128_bf16<<<gblocks(NT), blk, 0, stream>>>(sp,    D, nullptr, W_ih + 2 * WC, D, 1, n1buf, NT, 0);
    gemm128_bf16<<<gblocks(NT), blk, 0, stream>>>(fused, D, nullptr, W_hh + 2 * WC, D, 1, n2buf, NT, 0);

    // ---- final GRU combine + relu (all 302k rows share ordering) ----
    gru_final<<<(totAll + EB - 1) / EB, EB, 0, stream>>>(rbuf, zbuf, n1buf, n2buf, fused,
                                                         b_ih, b_hh, out, totAll);
}
